// PolicyEvaluationLayer_51290499448963
// MI455X (gfx1250) — matrix operand order repaired and checked by reading, not compiled, not run on hardware
//
#include <hip/hip_runtime.h>
#include <hip/hip_bf16.h>

typedef __attribute__((ext_vector_type(16))) _Float16 v16h;
typedef __attribute__((ext_vector_type(8)))  _Float16 v8h;
typedef __attribute__((ext_vector_type(8)))  float    v8f;

#define NS_  4096
#define NA_  16
#define NSA_ (NS_ * NA_)
#define BETA_ 0.99f

#if defined(__HIP_DEVICE_COMPILE__)
typedef __attribute__((__vector_size__(16))) int i32x4;
typedef __attribute__((address_space(1))) i32x4 gi32x4;   // global (__device__) int4
typedef __attribute__((address_space(3))) i32x4 li32x4;   // LDS (__shared__) int4
#endif

// ---------------- init: rcur=r, qcur=q, acc=h0*r ----------------
__global__ void k_init(const float* __restrict__ r, const float* __restrict__ q,
                       const float* __restrict__ h,
                       float* __restrict__ rcur, float* __restrict__ qcur,
                       float* __restrict__ acc) {
  int i = blockIdx.x * blockDim.x + threadIdx.x;
  float h0 = h[0];
  float rv = r[i], qv = q[i];
  rcur[i] = rv;
  qcur[i] = qv;
  acc[i]  = h0 * rv;
}

// ---------------- one-time P (f32) -> P16 (f16), 8 elems/thread ----------------
__global__ void k_convert_f16(const float* __restrict__ P, _Float16* __restrict__ P16) {
  long long t = (long long)blockIdx.x * blockDim.x + threadIdx.x;
  long long base = t * 8;
  const float4* src = (const float4*)(P + base);
  float4 a = src[0], b = src[1];
  v8h o;
  o[0] = (_Float16)a.x; o[1] = (_Float16)a.y; o[2] = (_Float16)a.z; o[3] = (_Float16)a.w;
  o[4] = (_Float16)b.x; o[5] = (_Float16)b.y; o[6] = (_Float16)b.z; o[7] = (_Float16)b.w;
  *(v8h*)(P16 + base) = o;
}

// ---------------- w_r[s] = dot(Pi[s,:], rcur[s*16:+16]); same for q ----------------
// wr16 and wq16 are CONTIGUOUS (wq16 == wr16 + NS_) so the sweep can stage both
// with one contiguous async copy.
__global__ void k_w(const float* __restrict__ Pi, const float* __restrict__ rcur,
                    const float* __restrict__ qcur,
                    _Float16* __restrict__ wr16, _Float16* __restrict__ wq16) {
  int s = blockIdx.x * blockDim.x + threadIdx.x;
  const float4* pi = (const float4*)(Pi   + (size_t)s * NA_);
  const float4* rv = (const float4*)(rcur + (size_t)s * NA_);
  const float4* qv = (const float4*)(qcur + (size_t)s * NA_);
  float wr = 0.f, wq = 0.f;
#pragma unroll
  for (int j = 0; j < 4; j++) {
    float4 p = pi[j], a = rv[j], b = qv[j];
    wr += p.x * a.x + p.y * a.y + p.z * a.z + p.w * a.w;
    wq += p.x * b.x + p.y * b.y + p.z * b.z + p.w * b.w;
  }
  wr16[s] = (_Float16)wr;
  wq16[s] = (_Float16)wq;
}

// ---------------- fused 2-RHS mat-vec sweep via WMMA f32_16x16x32_f16 ----------------
// Each wave owns 16 rows of P; A = 16x32 f16 tile of P, B columns {0,1} = w_r,w_q.
// Branchless B: each lane owns a precomputed LDS address; lanes with column>=2
// read a 32-byte zero pad and never advance.
template <bool FROM_F32>
__global__ __launch_bounds__(256) void k_sweep(const void* __restrict__ Psrc,
                                               const _Float16* __restrict__ wcat,  // wr||wq, 2*NS_
                                               const float* __restrict__ h, int kidx,
                                               float* __restrict__ rcur,
                                               float* __restrict__ qcur,
                                               float* __restrict__ acc) {
  __shared__ _Float16 ldsw[2 * NS_ + 16];  // wr | wq | 32B zero pad

  // ---- stage wr||wq (16 KB contiguous) into LDS, async if available ----
#if defined(__HIP_DEVICE_COMPILE__) && __has_builtin(__builtin_amdgcn_global_load_async_to_lds_b128)
  for (int i = threadIdx.x; i < (2 * NS_ * 2) / 16; i += 256) {
    __builtin_amdgcn_global_load_async_to_lds_b128(
        (gi32x4*)(void*)((const char*)wcat + (size_t)i * 16),
        (li32x4*)(void*)((char*)ldsw + (size_t)i * 16), 0, 0);
  }
#if __has_builtin(__builtin_amdgcn_s_wait_asynccnt)
  __builtin_amdgcn_s_wait_asynccnt(0);
#else
  asm volatile("s_wait_asynccnt 0x0" ::: "memory");
#endif
#else
  {
    const uint4* src = (const uint4*)wcat;
    uint4* dst = (uint4*)ldsw;
    for (int i = threadIdx.x; i < (2 * NS_) / 8; i += 256) dst[i] = src[i];
  }
#endif
  if (threadIdx.x < 16) ldsw[2 * NS_ + threadIdx.x] = (_Float16)0;  // zero pad
  __syncthreads();

  const int lane = threadIdx.x & 31;
  const int wave = threadIdx.x >> 5;
  const int m    = lane & 15;                    // B column / A row-in-tile
  const int hh   = lane >> 4;                    // 0 or 1 (lane half)
  const int row0 = (blockIdx.x * 8 + wave) * 16; // 16 rows per wave
  const int row  = row0 + m;
  const int kA   = hh * 8;                       // A-layout K sub-offset (halfs)
  const int kB   = hh * 8;                       // B-layout K sub-offset (halfs)

  // per-lane LDS byte address for B fragment + per-lane increment (branchless)
  int bByte = (m == 0) ? (kB * 2)
            : (m == 1) ? ((NS_ + kB) * 2)
                       : (2 * NS_ * 2);          // zero pad
  const int bInc = (m < 2) ? 64 : 0;             // 32 halfs per k-chunk
  const char* ldsb = (const char*)ldsw;

  v8f c = {};
  for (int k0 = 0; k0 < NS_; k0 += 128) {        // 4 WMMAs per group
    if constexpr (FROM_F32) {
      __builtin_prefetch((const float*)Psrc + (size_t)row * NS_ + k0 + 1024, 0, 0);
    } else {
      __builtin_prefetch((const _Float16*)Psrc + (size_t)row * NS_ + k0 + 2048, 0, 0);
    }
#pragma unroll
    for (int kk = 0; kk < 128; kk += 32) {
      const int kc = k0 + kk;
      // ---- A: lanes 0-15 hold K {0..7,16..23}, lanes 16-31 hold K {8..15,24..31}
      union { v16h v; v8h p[2]; } A;
      if constexpr (FROM_F32) {
        const float* Pr = (const float*)Psrc + (size_t)row * NS_ + kc;
        const float4* p0 = (const float4*)(Pr + kA);
        const float4* p1 = (const float4*)(Pr + 16 + kA);
        float4 x0 = p0[0], x1 = p0[1], y0 = p1[0], y1 = p1[1];
        v8h lo, hi;
        lo[0]=(_Float16)x0.x; lo[1]=(_Float16)x0.y; lo[2]=(_Float16)x0.z; lo[3]=(_Float16)x0.w;
        lo[4]=(_Float16)x1.x; lo[5]=(_Float16)x1.y; lo[6]=(_Float16)x1.z; lo[7]=(_Float16)x1.w;
        hi[0]=(_Float16)y0.x; hi[1]=(_Float16)y0.y; hi[2]=(_Float16)y0.z; hi[3]=(_Float16)y0.w;
        hi[4]=(_Float16)y1.x; hi[5]=(_Float16)y1.y; hi[6]=(_Float16)y1.z; hi[7]=(_Float16)y1.w;
        A.p[0] = lo; A.p[1] = hi;
      } else {
        const _Float16* Pr = (const _Float16*)Psrc + (size_t)row * NS_ + kc;
        A.p[0] = *(const v8h*)(Pr + kA);
        A.p[1] = *(const v8h*)(Pr + 16 + kA);
      }

      // ---- B: branchless ds_load_b128 x2 (zero lanes hit the pad)
      union { v16h v; v8h p[2]; } B;
      B.p[0] = *(const v8h*)(ldsb + bByte);
      B.p[1] = *(const v8h*)(ldsb + bByte + (bInc ? 32 : 16));
      bByte += bInc;

      c = __builtin_amdgcn_wmma_f32_16x16x32_f16(false, A.v, false, B.v,
                                                 (short)0, c, false, false);
    }
  }

  // ---- writeback: c[j] = D[M = j + 8*hh, N = m]; rows = row0 + 8*hh + j
  if (m < 2) {
    float hk = h[kidx];
    int rbase = row0 + hh * 8;
#pragma unroll
    for (int j = 0; j < 8; j++) {
      float d = c[j];
      int rr = rbase + j;
      if (m == 0) {
        rcur[rr] = d;
        acc[rr] += hk * d;   // q_prime += h_k * pv^k(r)
      } else {
        qcur[rr] = d;
      }
    }
  }
}

// ---------------- out = acc + BETA * h[10] * pv^9(q) ----------------
__global__ void k_final(const float* __restrict__ acc, const float* __restrict__ qcur,
                        const float* __restrict__ h, float* __restrict__ out) {
  int i = blockIdx.x * blockDim.x + threadIdx.x;
  out[i] = acc[i] + BETA_ * h[10] * qcur[i];
}

extern "C" void kernel_launch(void* const* d_in, const int* in_sizes, int n_in,
                              void* d_out, int out_size, void* d_ws, size_t ws_size,
                              hipStream_t stream) {
  const float* q  = (const float*)d_in[0];
  const float* Pi = (const float*)d_in[1];
  const float* P  = (const float*)d_in[2];
  const float* r  = (const float*)d_in[3];
  const float* h  = (const float*)d_in[4];
  float* out = (float*)d_out;
  (void)in_sizes; (void)n_in; (void)out_size;

  // workspace layout
  float*    rcur = (float*)d_ws;
  float*    qcur = rcur + NSA_;
  float*    acc  = qcur + NSA_;
  _Float16* wr16 = (_Float16*)(acc + NSA_);
  _Float16* wq16 = wr16 + NS_;          // contiguous with wr16
  _Float16* P16  = wq16 + NS_;

  size_t fixed    = (size_t)3 * NSA_ * sizeof(float) + (size_t)2 * NS_ * sizeof(_Float16);
  size_t p16Bytes = (size_t)NSA_ * NS_ * sizeof(_Float16);  // 512 MB
  bool fast = ws_size >= fixed + p16Bytes;

  k_init<<<NSA_ / 256, 256, 0, stream>>>(r, q, h, rcur, qcur, acc);
  if (fast) {
    long long nThreads = (long long)NSA_ * NS_ / 8;
    k_convert_f16<<<(int)(nThreads / 256), 256, 0, stream>>>(P, P16);
  }
  for (int k = 1; k <= 9; k++) {
    k_w<<<NS_ / 256, 256, 0, stream>>>(Pi, rcur, qcur, wr16, wq16);
    if (fast)
      k_sweep<false><<<NSA_ / 128, 256, 0, stream>>>((const void*)P16, wr16, h, k,
                                                     rcur, qcur, acc);
    else
      k_sweep<true><<<NSA_ / 128, 256, 0, stream>>>((const void*)P, wr16, h, k,
                                                    rcur, qcur, acc);
  }
  k_final<<<NSA_ / 256, 256, 0, stream>>>(acc, qcur, h, out);
}
